// GLeMaNet_9397388443815
// MI455X (gfx1250) — compile-verified, matrix operand order checked by reading, not run on hardware
//
#include <hip/hip_runtime.h>
#include <hip/hip_bf16.h>
#include <math.h>

#define BATCH 16
#define NNODE 1024
#define DIM   128
#define MTOT  (BATCH*NNODE)   // 16384
#define LDP   40              // padded LDS pitch (bf16 elems): 64B row + 16B pad
#define SMEM_BYTES (2 * 128 * LDP * 2)   // As + Bs, bf16

typedef __attribute__((ext_vector_type(16))) __bf16 v16bf;
typedef __attribute__((ext_vector_type(8)))  __bf16 v8bf;
typedef __attribute__((ext_vector_type(8)))  float  v8f;
typedef __attribute__((ext_vector_type(4)))  unsigned int v4u;
typedef __attribute__((ext_vector_type(8)))  unsigned int v8u;

// ---------------------------------------------------------------------------
// Tiled WMMA GEMM: C[M,N] = A[M,K] * B[K,N]   (bf16 inputs, f32 accumulate)
//   AT_BF16: A is bf16 in memory (else f32, converted while staging)
//   BT:      read B element (k,n) at B[n*LDB + k]
//   MODE:    0 plain | 1 +bias[col] | 2 relu | 3 hop: c*h + (1-c)*relu(acc)
//   TDM_A:   stage the (bf16) A tile with the Tensor Data Mover, LDS-pad to LDP
// Block: 256 threads (8 waves), tile 128x128, K-step 32. All dims divide tiles.
// Dynamic LDS only => As at LDS offset 0, Bs at 128*LDP*2 (needed for the D#).
// ---------------------------------------------------------------------------
template<int AT_BF16, int BT, int MODE, int LDA, int LDB, int LDC, int TDM_A>
__global__ __launch_bounds__(256) void gemm_wmma(
    const void* __restrict__ Aop, const float* __restrict__ Bop,
    float* __restrict__ Cop,
    int K,
    long sA, long sB, long sC,
    const float* __restrict__ bias,
    const float* __restrict__ coeff, long sCf,
    const float* __restrict__ Hmat, long sH)
{
  extern __shared__ __align__(16) __bf16 smem[];
  __bf16* As = smem;                 // LDS byte offset 0
  __bf16* Bs = smem + 128 * LDP;     // LDS byte offset 10240

  const int tid = threadIdx.x;
  const int m0 = blockIdx.y * 128;
  const int n0 = blockIdx.x * 128;
  const long bz = blockIdx.z;

  const __bf16* A16 = (const __bf16*)Aop + bz * sA;
  const float*  A32 = (const float*) Aop + bz * sA;
  const float*  Bp  = Bop + bz * sB;
  float*        Cp  = Cop + bz * sC;
  const float*  cfp = (MODE == 3) ? (coeff + bz * sCf) : nullptr;
  const float*  Hp  = (MODE == 3) ? (Hmat  + bz * sH)  : nullptr;

  const int lane = tid & 31, w = tid >> 5;
  const int wm = (w & 3) * 32;        // wave row offset within tile
  const int wn = (w >> 2) * 64;       // wave col offset within tile
  const int lr = lane & 15, lh = lane >> 4;

  // staging indices (each thread moves 16 elements per tile per operand)
  const int ar  = tid >> 1, ac = (tid & 1) << 4;          // A / B(BT) rows
  const int bkN = tid >> 3, bn0N = (tid & 7) << 4;        // B(non-BT)

  v8f acc[2][4];
#pragma unroll
  for (int i = 0; i < 2; ++i)
#pragma unroll
    for (int j = 0; j < 4; ++j)
#pragma unroll
      for (int e = 0; e < 8; ++e) acc[i][j][e] = 0.0f;

  for (int k0 = 0; k0 < K; k0 += 32) {
    // ---- stage A tile (128 x 32) ----
    if (TDM_A) {
      if (w == 0) {
        // Tensor DMA: 128 rows x 32 bf16 from row-stride-LDA matrix into LDS,
        // padding 4 DWORDs after every 16 DWORDs => LDP=40 bf16 pitch.
        unsigned long long ga = (unsigned long long)(uintptr_t)A16 +
                                (((unsigned long long)m0 * LDA + (unsigned)k0) << 1);
        v4u g0;
        g0[0] = 1u;                                       // count=1, user mode
        g0[1] = 0u;                                       // lds_addr = 0 (As)
        g0[2] = (unsigned)ga;                             // global_addr[31:0]
        g0[3] = (unsigned)((ga >> 32) & 0x1FFFFFFu) | (2u << 30);  // addr[56:32] | type=2
        v8u g1;
        g1[0] = (1u << 16)        // data_size = 2 bytes
              | (1u << 20)        // pad_enable
              | (3u << 22)        // pad_interval: 16 DWORDs
              | (3u << 25);       // pad_amount:   4 DWORDs
        g1[1] = ((unsigned)LDA & 0xFFFFu) << 16;          // tensor_dim0 lo16
        g1[2] = (((unsigned)LDA >> 16) & 0xFFFFu)         // tensor_dim0 hi16
              | ((unsigned)NNODE << 16);                  // tensor_dim1 lo16
        g1[3] = (32u << 16);                              // tile_dim0 = 32
        g1[4] = 128u;                                     // tile_dim1 = 128, tile_dim2 = 0
        g1[5] = (unsigned)LDA;                            // tensor_dim0_stride lo32
        g1[6] = 0u;
        g1[7] = 0u;
        asm volatile("tensor_load_to_lds %0, %1" :: "s"(g0), "s"(g1) : "memory");
      }
    } else {
      __bf16* dst = &As[ar * LDP + ac];
      if (AT_BF16) {
        const __bf16* src = A16 + (size_t)(m0 + ar) * LDA + k0 + ac;
        *(v8bf*)dst       = *(const v8bf*)src;
        *(v8bf*)(dst + 8) = *(const v8bf*)(src + 8);
      } else {
        const float* src = A32 + (size_t)(m0 + ar) * LDA + k0 + ac;
        v8bf d0, d1;
#pragma unroll
        for (int i = 0; i < 8; ++i) { d0[i] = (__bf16)src[i]; d1[i] = (__bf16)src[8 + i]; }
        *(v8bf*)dst = d0; *(v8bf*)(dst + 8) = d1;
        if (k0 + 32 < K) __builtin_prefetch(src + 32, 0, 3);   // global_prefetch_b8
      }
    }
    // ---- stage B tile (32 x 128), stored N-major: Bs[n][k] ----
    if (BT) {
      const float* src = Bp + (size_t)(n0 + ar) * LDB + k0 + ac;
      v8bf d0, d1;
#pragma unroll
      for (int i = 0; i < 8; ++i) { d0[i] = (__bf16)src[i]; d1[i] = (__bf16)src[8 + i]; }
      __bf16* dst = &Bs[ar * LDP + ac];
      *(v8bf*)dst = d0; *(v8bf*)(dst + 8) = d1;
      if (k0 + 32 < K) __builtin_prefetch(src + 32, 0, 3);
    } else {
      const float* src = Bp + (size_t)(k0 + bkN) * LDB + n0 + bn0N;
#pragma unroll
      for (int j = 0; j < 16; ++j) Bs[(bn0N + j) * LDP + bkN] = (__bf16)src[j];
      if (k0 + 32 < K) __builtin_prefetch(src + 32 * LDB, 0, 3);
    }
    if (TDM_A && w == 0) __builtin_amdgcn_s_wait_tensorcnt(0);
    __syncthreads();

    // ---- fragments (ISA 16-bit A 16x32 layout; B tile is N-major so same shape) ----
    v16bf af[2];
#pragma unroll
    for (int fm = 0; fm < 2; ++fm) {
      const __bf16* p = &As[(wm + fm * 16 + lr) * LDP];
      v8bf lo = *(const v8bf*)(p + (lh << 3));
      v8bf hi = *(const v8bf*)(p + 16 + (lh << 3));
      af[fm] = __builtin_shufflevector(lo, hi, 0,1,2,3,4,5,6,7,8,9,10,11,12,13,14,15);
    }
    v16bf bf_[4];
#pragma unroll
    for (int fn = 0; fn < 4; ++fn) {
      const __bf16* p = &Bs[(wn + fn * 16 + lr) * LDP];
      v8bf lo = *(const v8bf*)(p + (lh << 3));
      v8bf hi = *(const v8bf*)(p + 16 + (lh << 3));
      bf_[fn] = __builtin_shufflevector(lo, hi, 0,1,2,3,4,5,6,7,8,9,10,11,12,13,14,15);
    }

#pragma unroll
    for (int fm = 0; fm < 2; ++fm)
#pragma unroll
      for (int fn = 0; fn < 4; ++fn)
        acc[fm][fn] = __builtin_amdgcn_wmma_f32_16x16x32_bf16(
            false, af[fm], false, bf_[fn], (short)0, acc[fm][fn], false, false);

    __syncthreads();
  }

  // ---- epilogue (C layout: VGPR r -> row r + 8*lh, col = lr); LDC is
  //      compile-time so row strides become immediate store offsets ----
#pragma unroll
  for (int fm = 0; fm < 2; ++fm) {
#pragma unroll
    for (int fn = 0; fn < 4; ++fn) {
      const int rbase = m0 + wm + fm * 16 + lh * 8;
      const int col   = n0 + wn + fn * 16 + lr;
      float* cb = Cp + (size_t)rbase * LDC + col;
      const float* hb = (MODE == 3) ? (Hp + (size_t)rbase * LDC + col) : nullptr;
      const float* cfb = (MODE == 3) ? (cfp + rbase) : nullptr;
      const float bv = (MODE == 1) ? bias[col] : 0.0f;
#pragma unroll
      for (int r = 0; r < 8; ++r) {
        float v = acc[fm][fn][r];
        if (MODE == 1) v += bv;
        if (MODE == 2) v = fmaxf(v, 0.0f);
        if (MODE == 3) {
          const float cf = cfb[r];
          v = cf * hb[r * LDC] + (1.0f - cf) * fmaxf(v, 0.0f);
        }
        cb[r * LDC] = v;
      }
    }
  }
}

// ---------------------------------------------------------------------------
// Column softmax: att[:,j,k] = softmax_j(where(adj>0, e+e^T, 0)) * adj, bf16 out
// One thread per column k; lanes span consecutive k -> coalesced e/adj reads.
// ---------------------------------------------------------------------------
__global__ __launch_bounds__(256) void softmax_col(const float* __restrict__ e,
                                                   const float* __restrict__ adj,
                                                   __bf16* __restrict__ att)
{
  const int b = blockIdx.y;
  const int k = blockIdx.x * 256 + threadIdx.x;
  const float* eb = e   + (size_t)b * NNODE * NNODE;
  const float* ab = adj + (size_t)b * NNODE * NNODE;
  __bf16*      ob = att + (size_t)b * NNODE * NNODE;

  float m = -3.4e38f;
  for (int j = 0; j < NNODE; ++j) {
    float a = ab[(size_t)j * NNODE + k];
    float v = (a > 0.0f) ? (eb[(size_t)j * NNODE + k] + eb[(size_t)k * NNODE + j]) : 0.0f;
    m = fmaxf(m, v);
  }
  float s = 0.0f;
  for (int j = 0; j < NNODE; ++j) {
    float a = ab[(size_t)j * NNODE + k];
    float v = (a > 0.0f) ? (eb[(size_t)j * NNODE + k] + eb[(size_t)k * NNODE + j]) : 0.0f;
    s += __expf(v - m);
  }
  const float inv = 1.0f / s;
  for (int j = 0; j < NNODE; ++j) {
    float a = ab[(size_t)j * NNODE + k];
    float v = (a > 0.0f) ? (eb[(size_t)j * NNODE + k] + eb[(size_t)k * NNODE + j]) : 0.0f;
    ob[(size_t)j * NNODE + k] = (__bf16)(__expf(v - m) * inv * a);
  }
}

// coeff = sigmoid([h, az] . gw + gb), one wave32 per node
__global__ __launch_bounds__(256) void gate_kernel(const float* __restrict__ h,
                                                   const float* __restrict__ az,
                                                   const float* __restrict__ gw,
                                                   const float* __restrict__ gbp,
                                                   float* __restrict__ coeff)
{
  const int node = blockIdx.x * 8 + (threadIdx.x >> 5);
  const int lane = threadIdx.x & 31;
  const float* hp = h  + (size_t)node * DIM;
  const float* ap = az + (size_t)node * DIM;
  float s = 0.0f;
#pragma unroll
  for (int i = lane; i < DIM; i += 32) s += hp[i] * gw[i] + ap[i] * gw[DIM + i];
#pragma unroll
  for (int off = 16; off > 0; off >>= 1) s += __shfl_xor(s, off, 32);
  if (lane == 0) coeff[node] = 1.0f / (1.0f + __expf(-(s + gbp[0])));
}

__global__ __launch_bounds__(256) void sub_kernel(const float4* __restrict__ a,
                                                  const float4* __restrict__ b,
                                                  float4* __restrict__ c, int n)
{
  const int i = blockIdx.x * blockDim.x + threadIdx.x;
  if (i < n) {
    float4 x = a[i], y = b[i];
    c[i] = make_float4(x.x - y.x, x.y - y.y, x.z - y.z, x.w - y.w);
  }
}

__global__ __launch_bounds__(128) void pool_kernel(const float* __restrict__ h,
                                                   const float* __restrict__ valid,
                                                   float* __restrict__ pooled)
{
  const int b = blockIdx.x, d = threadIdx.x;
  const float* hb = h + (size_t)b * NNODE * DIM;
  const float* vb = valid + (size_t)b * NNODE;
  float s = 0.0f, vs = 0.0f;
  for (int i = 0; i < NNODE; ++i) {
    float v = vb[i];
    s  += hb[(size_t)i * DIM + d] * v;
    vs += v;
  }
  pooled[b * DIM + d] = s / vs;
}

__global__ __launch_bounds__(128) void fc_chain(const float* __restrict__ pooled,
    const float* __restrict__ w0, const float* __restrict__ b0,
    const float* __restrict__ w1, const float* __restrict__ b1,
    const float* __restrict__ w2, const float* __restrict__ b2,
    const float* __restrict__ w3, const float* __restrict__ b3,
    float* __restrict__ out)
{
  __shared__ float y[DIM], yn[DIM], red[DIM];
  const int b = blockIdx.x, d = threadIdx.x;
  y[d] = pooled[b * DIM + d];
  __syncthreads();
  const float* Ws[3] = { w0, w1, w2 };
  const float* Bs[3] = { b0, b1, b2 };
  for (int l = 0; l < 3; ++l) {
    float s = 0.0f;
    const float* wr = Ws[l] + (size_t)d * DIM;
    for (int e = 0; e < DIM; ++e) s += wr[e] * y[e];
    yn[d] = fmaxf(s + Bs[l][d], 0.0f);
    __syncthreads();
    y[d] = yn[d];
    __syncthreads();
  }
  red[d] = y[d] * w3[d];
  __syncthreads();
  for (int off = 64; off > 0; off >>= 1) {
    if (d < off) red[d] += red[d + off];
    __syncthreads();
  }
  if (d == 0) out[b] = 1.0f / (1.0f + __expf(-(red[0] + b3[0])));
}

// ---------------------------------------------------------------------------
extern "C" void kernel_launch(void* const* d_in, const int* in_sizes, int n_in,
                              void* d_out, int out_size, void* d_ws, size_t ws_size,
                              hipStream_t stream)
{
  const float* x        = (const float*)d_in[0];
  const float* adj1     = (const float*)d_in[1];
  const float* adj2     = (const float*)d_in[2];
  const float* valid    = (const float*)d_in[3];
  const float* embede_w = (const float*)d_in[4];
  const float* gW       = (const float*)d_in[5];
  const float* gb       = (const float*)d_in[6];
  const float* gA       = (const float*)d_in[7];
  const float* gate_w   = (const float*)d_in[8];
  const float* gate_b   = (const float*)d_in[9];
  const float* fc0_w = (const float*)d_in[10]; const float* fc0_b = (const float*)d_in[11];
  const float* fc1_w = (const float*)d_in[12]; const float* fc1_b = (const float*)d_in[13];
  const float* fc2_w = (const float*)d_in[14]; const float* fc2_b = (const float*)d_in[15];
  const float* fc3_w = (const float*)d_in[16]; const float* fc3_b = (const float*)d_in[17];
  float* out = (float*)d_out;

  // ---- workspace layout (e region reused for az / z ping-pong buffers) ----
  char* ws = (char*)d_ws;
  const size_t HB = (size_t)MTOT * DIM * sizeof(float);        // 8 MiB
  float* hcur = (float*)(ws + 0 * HB);
  float* hl   = (float*)(ws + 1 * HB);
  float* tmat = (float*)(ws + 2 * HB);
  float* emat = (float*)(ws + 3 * HB);                         // 64 MiB (3HB..11HB)
  float* az   = emat;                                          // reuse (e dead)
  float* zbuf[2][2] = { { (float*)(ws + 4 * HB), (float*)(ws + 5 * HB) },
                        { (float*)(ws + 6 * HB), (float*)(ws + 7 * HB) } };
  __bf16* att[2];
  att[0] = (__bf16*)(ws + 11 * HB);
  att[1] = att[0] + (size_t)BATCH * NNODE * NNODE;
  float* coeff[2];
  coeff[0] = (float*)(att[1] + (size_t)BATCH * NNODE * NNODE);
  coeff[1] = coeff[0] + MTOT;
  float* pooled = coeff[1] + MTOT;

  const dim3 blk(256);
  const dim3 gEmb(1, MTOT / 128, 1);
  const dim3 gE(NNODE / 128, NNODE / 128, BATCH);
  const dim3 gHop(1, NNODE / 128, BATCH);

  // embed: hcur = x @ embede_w^T  (no bias)
  gemm_wmma<0, 1, 0, 128, 128, 128, 0><<<gEmb, blk, SMEM_BYTES, stream>>>(
      x, embede_w, hcur, 128, 0, 0, 0, nullptr, nullptr, 0, nullptr, 0);

  for (int k = 0; k < 4; ++k) {
    const float* Wk  = gW + (size_t)k * DIM * DIM;
    const float* bk  = gb + (size_t)k * DIM;
    const float* Ak  = gA + (size_t)k * DIM * DIM;
    const float* gwk = gate_w + (size_t)k * 2 * DIM;
    const float* gbk = gate_b + k;

    // hl = hcur @ Wk^T + bk
    gemm_wmma<0, 1, 1, 128, 128, 128, 0><<<gEmb, blk, SMEM_BYTES, stream>>>(
        hcur, Wk, hl, 128, 0, 0, 0, bk, nullptr, 0, nullptr, 0);
    // tmat = hl @ Ak
    gemm_wmma<0, 0, 0, 128, 128, 128, 0><<<gEmb, blk, SMEM_BYTES, stream>>>(
        hl, Ak, tmat, 128, 0, 0, 0, nullptr, nullptr, 0, nullptr, 0);
    // e = tmat @ hl^T   (batched 1024x1024x128)
    gemm_wmma<0, 1, 0, 128, 128, NNODE, 0><<<gE, blk, SMEM_BYTES, stream>>>(
        tmat, hl, emat, 128,
        (long)NNODE * DIM, (long)NNODE * DIM, (long)NNODE * NNODE,
        nullptr, nullptr, 0, nullptr, 0);
    // column softmax -> att (bf16), both branches
    softmax_col<<<dim3(NNODE / 256, BATCH), blk, 0, stream>>>(emat, adj1, att[0]);
    softmax_col<<<dim3(NNODE / 256, BATCH), blk, 0, stream>>>(emat, adj2, att[1]);

    float* zfin[2];
    for (int br = 0; br < 2; ++br) {
      // az = relu(att @ hl)   (A tile staged by the Tensor Data Mover)
      gemm_wmma<1, 0, 2, NNODE, 128, 128, 1><<<gHop, blk, SMEM_BYTES, stream>>>(
          att[br], hl, az, NNODE,
          (long)NNODE * NNODE, (long)NNODE * DIM, (long)NNODE * DIM,
          nullptr, nullptr, 0, nullptr, 0);
      // coeff = sigmoid([hl, az] . gw + gb)
      gate_kernel<<<MTOT / 8, blk, 0, stream>>>(hl, az, gwk, gbk, coeff[br]);
      // nhop = k+1 fused hop iterations: z = c*hl + (1-c)*relu(att @ z)
      float* zsrc = hl;
      for (int hh = 0; hh <= k; ++hh) {
        float* zdst = zbuf[br][hh & 1];
        gemm_wmma<1, 0, 3, NNODE, 128, 128, 1><<<gHop, blk, SMEM_BYTES, stream>>>(
            att[br], zsrc, zdst, NNODE,
            (long)NNODE * NNODE, (long)NNODE * DIM, (long)NNODE * DIM,
            nullptr, coeff[br], NNODE, hl, (long)NNODE * DIM);
        zsrc = zdst;
      }
      zfin[br] = zsrc;
    }
    // hcur = z(adj2) - z(adj1)
    sub_kernel<<<(MTOT * DIM / 4) / 256, blk, 0, stream>>>(
        (const float4*)zfin[1], (const float4*)zfin[0], (float4*)hcur, MTOT * DIM / 4);
  }

  pool_kernel<<<BATCH, DIM, 0, stream>>>(hcur, valid, pooled);
  fc_chain<<<BATCH, DIM, 0, stream>>>(pooled, fc0_w, fc0_b, fc1_w, fc1_b,
                                      fc2_w, fc2_b, fc3_w, fc3_b, out);
}
